// Model_61074434949741
// MI455X (gfx1250) — matrix operand order repaired and checked by reading, not compiled, not run on hardware
//
#include <hip/hip_runtime.h>
#include <math.h>

// ---------------------------------------------------------------------------
// MoE-as-GEMM for MI455X (gfx1250, wave32). Five of six layers (all but gate) are bf16 WMMA
// GEMMs with f32 accumulation. Expert mixing = per-expert partial accumulators
// folded with gate[row,e] at expert boundaries, so the inner loop moves PURE
// bf16 tiles. TDM (tensor_load_to_lds) double-buffer pipeline: prefetch tile
// t+1 into the alternate LDS buffer while 8x v_wmma_f32_16x16x32_bf16 consume
// tile t. Hardware D# padding gives an 80B LDS row stride for conflict-free
// ds_load_b128 fragment loads. Compute-bound: ~214 GFLOP vs ~60MB bytes.
// ---------------------------------------------------------------------------

#define BM 128
#define BN 128
#define BK 32
#define LDSTRIDE 40   // shorts per LDS tile row: 64B data + 16B pad (TDM pad feature)

#define USE_TDM (__has_builtin(__builtin_amdgcn_tensor_load_to_lds))

typedef __bf16 bf16x16 __attribute__((ext_vector_type(16)));
typedef __bf16 bf16x8  __attribute__((ext_vector_type(8)));
typedef float  v8f     __attribute__((ext_vector_type(8)));
typedef float  f4_t    __attribute__((ext_vector_type(4)));
typedef unsigned int   u32x4  __attribute__((ext_vector_type(4)));
typedef int            i32x4v __attribute__((ext_vector_type(4)));
typedef int            i32x8v __attribute__((ext_vector_type(8)));
typedef unsigned short u16x8  __attribute__((ext_vector_type(8)));

__device__ __forceinline__ unsigned short f2bf(float f) {
    union { float f; unsigned int u; } v; v.f = f;
    unsigned int r = v.u + 0x7FFFu + ((v.u >> 16) & 1u);   // RNE
    return (unsigned short)(r >> 16);
}
__device__ __forceinline__ float bf2f(unsigned short u) {
    union { unsigned int u; float f; } v; v.u = ((unsigned int)u) << 16;
    return v.f;
}
__device__ __forceinline__ bf16x16 frag16(const unsigned short* lo, const unsigned short* hi) {
    bf16x8 a = *(const bf16x8*)lo;
    bf16x8 b = *(const bf16x8*)hi;
    return __builtin_shufflevector(a, b, 0,1,2,3,4,5,6,7,8,9,10,11,12,13,14,15);
}

#if USE_TDM
// 2D bf16 tile DMA: 128 rows x 32 elements, row stride rowStride elements,
// zero-fill rows >= rowsRem. LDS dest gets 16B pad after every 64B row
// (pad_interval=3 -> 16 DWORDs, pad_amount=3 -> 4 DWORDs) => 80B LDS stride.
__device__ __forceinline__ void tdm_tile_load(unsigned ldsAddr,
                                              const unsigned short* gptr,
                                              long rowStride, int rowsRem)
{
    unsigned long long ga = (unsigned long long)gptr;
    u32x4 g0;
    g0[0] = 1u;                                             // count=1 (valid descriptor)
    g0[1] = ldsAddr;                                        // lds_addr
    g0[2] = (unsigned)ga;                                   // global_addr[31:0]
    g0[3] = ((unsigned)(ga >> 32) & 0x01FFFFFFu) | (2u << 30); // global_addr[56:32] | type=2
    i32x8v g1;
    g1[0] = (int)((1u << 16)    // data_size = 2 bytes
                | (1u << 20)    // pad_enable
                | (3u << 22)    // pad_interval: 16 DWORDs (64B)
                | (3u << 25));  // pad_amount: 4 DWORDs (16B)
    g1[1] = (int)(32u << 16);                               // tensor_dim0 = 32 (lo16)
    g1[2] = (int)(((unsigned)rowsRem & 0xFFFFu) << 16);     // dim0 hi | tensor_dim1 lo16
    g1[3] = (int)((((unsigned)rowsRem >> 16) & 0xFFFFu) | (32u << 16)); // dim1 hi | tile_dim0=32
    g1[4] = BM;                                             // tile_dim1=128, tile_dim2=0
    g1[5] = (int)(unsigned)rowStride;                       // tensor_dim0_stride lo32
    g1[6] = (int)(((unsigned long long)rowStride >> 32) & 0xFFFFu);
    g1[7] = 0;
    i32x4v z4 = {0, 0, 0, 0};
    i32x8v z8 = {0, 0, 0, 0, 0, 0, 0, 0};
    // 6-arg form on this toolchain: (g0, g1, g2, g3, extra_group, cpol)
    __builtin_amdgcn_tensor_load_to_lds(g0, g1, z4, z4, z8, 0);
}
#endif

// C[M,N] = op( sum_e gate[b,e] * (Abf[M,KE] x Bt_e) + bias )
// Bt is bf16 transposed weights [N x Kfull]; expert e occupies cols [e*KE,(e+1)*KE).
template<bool EXPERT, bool ELU, bool OUTBF16>
__global__ __launch_bounds__(256)
void gemm_wmma_bf16(const unsigned short* __restrict__ Abf,   // [M x KE] bf16
                    const float* __restrict__ gatep,          // [M x 8] (EXPERT)
                    const unsigned short* __restrict__ Bt,    // [N x Kfull] bf16
                    const float* __restrict__ biasp,          // [N] or [8 x N]
                    void* __restrict__ Cout,
                    int M, int N, int KE, int Kfull)
{
    __shared__ unsigned short As[2][BM * LDSTRIDE];
    __shared__ unsigned short Bs[2][BN * LDSTRIDE];
    __shared__ float Gs[BM * 8];

    const int tid   = threadIdx.x;
    const int lane  = tid & 31;
    const int wave  = tid >> 5;
    const int wm    = wave & 3;     // 4 waves along M (32-row strips)
    const int wn    = wave >> 2;    // 2 waves along N (64-col strips)
    const int laneN = lane & 15;
    const int half  = lane >> 4;

    const int m0 = blockIdx.y * BM;
    const int n0 = blockIdx.x * BN;
    const int NE = EXPERT ? 8 : 1;
    const int rowsRemB = (N - n0 < BN) ? (N - n0) : BN;

    if (EXPERT) {   // stage gate tile: 128 rows x 8, contiguous 1024 floats
        ((f4_t*)Gs)[tid] = ((const f4_t*)(gatep + (size_t)m0 * 8))[tid];
    }

    v8f facc[2][4];
    v8f acc[2][4];
    #pragma unroll
    for (int mi = 0; mi < 2; mi++)
        #pragma unroll
        for (int ni = 0; ni < 4; ni++)
            #pragma unroll
            for (int t = 0; t < 8; t++) { facc[mi][ni][t] = 0.0f; acc[mi][ni][t] = 0.0f; }

    // per-tile compute: 2 A frags + 4 B frags -> 8 WMMAs into acc
    auto computeTile = [&](const unsigned short* Ab, const unsigned short* Bb) {
        // A 16x32 frag: lane(row=laneN, half): K = {8h..8h+7} ++ {16+8h..23+8h}
        bf16x16 afr[2];
        #pragma unroll
        for (int mi = 0; mi < 2; mi++) {
            const unsigned short* basep = Ab + (wm * 32 + mi * 16 + laneN) * LDSTRIDE;
            afr[mi] = frag16(basep + 8 * half, basep + 16 + 8 * half);
        }
        // B 32x16 frag (LDS holds [n][k]): lane(col=laneN, half): K = {8h..8h+7} ++ {16+8h..23+8h}
        bf16x16 bfr[4];
        #pragma unroll
        for (int ni = 0; ni < 4; ni++) {
            const unsigned short* basep = Bb + (wn * 64 + ni * 16 + laneN) * LDSTRIDE + 8 * half;
            bfr[ni] = frag16(basep, basep + 16);
        }
        #pragma unroll
        for (int mi = 0; mi < 2; mi++)
            #pragma unroll
            for (int ni = 0; ni < 4; ni++)
                acc[mi][ni] = __builtin_amdgcn_wmma_f32_16x16x32_bf16(
                    false, afr[mi], false, bfr[ni], (short)0, acc[mi][ni], false, false);
    };

    const int tilesPerE = KE / BK;
    const int T = NE * tilesPerE;

#if USE_TDM
    // prime the pipeline: tile 0 -> buffer 0
    if (tid < 32) {
        tdm_tile_load((unsigned)(unsigned long long)&As[0][0],
                      Abf + (size_t)m0 * KE, (long)KE, BM);
        tdm_tile_load((unsigned)(unsigned long long)&Bs[0][0],
                      Bt + (size_t)n0 * Kfull, (long)Kfull, rowsRemB);
    }
    int i0_pf = BK, e_pf = 0;
    if (i0_pf >= KE) { i0_pf = 0; e_pf = 1; }
#endif

    int tileInE = 0;
    int e_c = 0;
    for (int t = 0; t < T; ++t) {
#if USE_TDM
        const int buf = t & 1;
        if (tid < 32)
            __builtin_amdgcn_s_wait_tensorcnt(0);   // tile t landed (wave 0's DMAs)
        __syncthreads();                            // publish LDS to all waves
        // prefetch tile t+1 into the buffer everyone just finished reading
        if ((t + 1 < T) && tid < 32) {
            tdm_tile_load((unsigned)(unsigned long long)&As[buf ^ 1][0],
                          Abf + (size_t)m0 * KE + i0_pf, (long)KE, BM);
            tdm_tile_load((unsigned)(unsigned long long)&Bs[buf ^ 1][0],
                          Bt + (size_t)n0 * Kfull + (size_t)e_pf * KE + i0_pf,
                          (long)Kfull, rowsRemB);
        }
        i0_pf += BK;
        if (i0_pf >= KE) { i0_pf = 0; ++e_pf; }

        computeTile(&As[buf][0], &Bs[buf][0]);      // overlaps with DMA of t+1
#else
        // fallback: plain b128 global->LDS copy (no conversion, padded stride)
        {
            const int i0 = tileInE * BK;
            #pragma unroll
            for (int it = 0; it < 2; ++it) {
                int idx = tid + it * 256;        // 0..511
                int row = idx >> 2;              // 0..127
                int c   = (idx & 3) * 8;         // short offset in row
                u16x8 v = *(const u16x8*)(Abf + (size_t)(m0 + row) * KE + i0 + c);
                *(u16x8*)(&As[0][row * LDSTRIDE + c]) = v;
            }
            #pragma unroll
            for (int it = 0; it < 2; ++it) {
                int idx = tid + it * 256;
                int row = idx >> 2;
                int c   = (idx & 3) * 8;
                u16x8 v;
                if (row < rowsRemB)
                    v = *(const u16x8*)(Bt + (size_t)(n0 + row) * Kfull + (size_t)e_c * KE + i0 + c);
                else {
                    #pragma unroll
                    for (int j = 0; j < 8; j++) v[j] = 0;
                }
                *(u16x8*)(&Bs[0][row * LDSTRIDE + c]) = v;
            }
        }
        __syncthreads();
        computeTile(&As[0][0], &Bs[0][0]);
        __syncthreads();
#endif

        // expert boundary: fold partial into final accumulator with gate[row,e]
        if (++tileInE == tilesPerE) {
            tileInE = 0;
            #pragma unroll
            for (int mi = 0; mi < 2; mi++) {
                #pragma unroll
                for (int i = 0; i < 8; i++) {
                    float g = 1.0f;
                    if (EXPERT) {
                        int rl = wm * 32 + mi * 16 + i + 8 * half;
                        g = Gs[rl * 8 + e_c];
                    }
                    #pragma unroll
                    for (int ni = 0; ni < 4; ni++) {
                        facc[mi][ni][i] += g * acc[mi][ni][i];
                        acc[mi][ni][i] = 0.0f;
                    }
                }
            }
            ++e_c;
        }
    }

    // ---- epilogue: bias (+ gate-blended expert bias), ELU, store ----
    #pragma unroll
    for (int mi = 0; mi < 2; mi++) {
        #pragma unroll
        for (int ni = 0; ni < 4; ni++) {
            int coln = n0 + wn * 64 + ni * 16 + laneN;
            if (coln < N) {
                float eb[8];
                float pb = 0.0f;
                if (EXPERT) {
                    #pragma unroll
                    for (int e2 = 0; e2 < 8; e2++) eb[e2] = biasp[(size_t)e2 * N + coln];
                } else {
                    pb = biasp[coln];
                }
                #pragma unroll
                for (int i = 0; i < 8; i++) {
                    int rl   = wm * 32 + mi * 16 + i + 8 * half;
                    int rowm = m0 + rl;                       // C layout: M = i + 8*half
                    float bias = pb;
                    if (EXPERT) {
                        #pragma unroll
                        for (int e2 = 0; e2 < 8; e2++) bias += Gs[rl * 8 + e2] * eb[e2];
                    }
                    float x = facc[mi][ni][i] + bias;
                    if (ELU) x = (x > 0.0f) ? x : (__expf(x) - 1.0f);
                    if (OUTBF16)
                        ((unsigned short*)Cout)[(size_t)rowm * N + coln] = f2bf(x);
                    else
                        ((float*)Cout)[(size_t)rowm * N + coln] = x;
                }
            }
        }
    }
}

// gate = h2(bf16) @ g_w3 + g_b3   (N=8, trivial)
__global__ __launch_bounds__(256)
void gate_kernel(const unsigned short* __restrict__ h, const float* __restrict__ w,
                 const float* __restrict__ bias, float* __restrict__ gate, int Brows)
{
    int idx = blockIdx.x * 256 + threadIdx.x;
    int b = idx >> 3;
    int e = idx & 7;
    if (b >= Brows) return;
    const unsigned short* hr = h + (size_t)b * 512;
    float s = bias[e];
    #pragma unroll 4
    for (int k = 0; k < 512; k++)
        s += bf2f(hr[k]) * w[k * 8 + e];
    gate[idx] = s;
}

// f32 -> bf16, 4 elements/thread
__global__ __launch_bounds__(256)
void cvt_bf16_kernel(const float* __restrict__ s, unsigned short* __restrict__ d, int n4)
{
    int i = blockIdx.x * 256 + threadIdx.x;
    if (i >= n4) return;
    f4_t v = ((const f4_t*)s)[i];
    unsigned long long p =
          (unsigned long long)f2bf(v[0])
        | ((unsigned long long)f2bf(v[1]) << 16)
        | ((unsigned long long)f2bf(v[2]) << 32)
        | ((unsigned long long)f2bf(v[3]) << 48);
    ((unsigned long long*)d)[i] = p;
}

// W[K][N] f32 -> Wt[N][K] bf16 (32x32 LDS tiles; K,N multiples of 32)
__global__ __launch_bounds__(256)
void transpose_bf16_kernel(const float* __restrict__ W, unsigned short* __restrict__ Wt,
                           int K, int N)
{
    __shared__ float t[32][33];
    int tx = threadIdx.x, ty = threadIdx.y;        // (32, 8)
    int n0 = blockIdx.x * 32, k0 = blockIdx.y * 32;
    #pragma unroll
    for (int j = 0; j < 4; j++)
        t[ty + j * 8][tx] = W[(size_t)(k0 + ty + j * 8) * N + n0 + tx];
    __syncthreads();
    #pragma unroll
    for (int j = 0; j < 4; j++)
        Wt[(size_t)(n0 + ty + j * 8) * K + k0 + tx] = f2bf(t[tx][ty + j * 8]);
}

extern "C" void kernel_launch(void* const* d_in, const int* in_sizes, int n_in,
                              void* d_out, int out_size, void* d_ws, size_t ws_size,
                              hipStream_t stream) {
    const float* X    = (const float*)d_in[0];
    const float* g_w1 = (const float*)d_in[1];
    const float* g_b1 = (const float*)d_in[2];
    const float* g_w2 = (const float*)d_in[3];
    const float* g_b2 = (const float*)d_in[4];
    const float* g_w3 = (const float*)d_in[5];
    const float* g_b3 = (const float*)d_in[6];
    const float* e_w1 = (const float*)d_in[7];    // [8,480,512] == stacked [3840,512]
    const float* e_b1 = (const float*)d_in[8];
    const float* e_w2 = (const float*)d_in[9];    // [4096,512]
    const float* e_b2 = (const float*)d_in[10];
    const float* e_w3 = (const float*)d_in[11];   // [4096,480]
    const float* e_b3 = (const float*)d_in[12];
    float* out = (float*)d_out;

    const int Mrows = 16384;

    // workspace carve-up (256B aligned)
    size_t off = 0;
    char* base = (char*)d_ws;
    auto alloc = [&](size_t bytes) -> void* {
        void* p = base + off;
        off += (bytes + 255) & ~(size_t)255;
        return p;
    };
    unsigned short* Xbf  = (unsigned short*)alloc((size_t)Mrows * 480 * 2);
    unsigned short* act0 = (unsigned short*)alloc((size_t)Mrows * 512 * 2);  // h1 -> mid1
    unsigned short* act1 = (unsigned short*)alloc((size_t)Mrows * 512 * 2);  // h2 -> mid2
    float*          gate = (float*)alloc((size_t)Mrows * 8 * 4);
    unsigned short* w1t  = (unsigned short*)alloc((size_t)512 * 480 * 2);
    unsigned short* w2t  = (unsigned short*)alloc((size_t)512 * 512 * 2);
    unsigned short* ew1t = (unsigned short*)alloc((size_t)512 * 3840 * 2);
    unsigned short* ew2t = (unsigned short*)alloc((size_t)512 * 4096 * 2);
    unsigned short* ew3t = (unsigned short*)alloc((size_t)480 * 4096 * 2);

    dim3 blk(256);
    dim3 tblk(32, 8);

    // pre-pass: bf16 copies of X and transposed bf16 weights
    cvt_bf16_kernel<<<(Mrows * 480 / 4 + 255) / 256, blk, 0, stream>>>(X, Xbf, Mrows * 480 / 4);
    transpose_bf16_kernel<<<dim3(512 / 32, 480 / 32),  tblk, 0, stream>>>(g_w1, w1t, 480, 512);
    transpose_bf16_kernel<<<dim3(512 / 32, 512 / 32),  tblk, 0, stream>>>(g_w2, w2t, 512, 512);
    transpose_bf16_kernel<<<dim3(512 / 32, 3840 / 32), tblk, 0, stream>>>(e_w1, ew1t, 3840, 512);
    transpose_bf16_kernel<<<dim3(512 / 32, 4096 / 32), tblk, 0, stream>>>(e_w2, ew2t, 4096, 512);
    transpose_bf16_kernel<<<dim3(480 / 32, 4096 / 32), tblk, 0, stream>>>(e_w3, ew3t, 4096, 480);

    dim3 g512(512 / BN, Mrows / BM);
    dim3 g480((480 + BN - 1) / BN, Mrows / BM);

    // gating MLP
    gemm_wmma_bf16<false, true,  true ><<<g512, blk, 0, stream>>>(Xbf,  nullptr, w1t,  g_b1, act0, Mrows, 512, 480, 480);
    gemm_wmma_bf16<false, true,  true ><<<g512, blk, 0, stream>>>(act0, nullptr, w2t,  g_b2, act1, Mrows, 512, 512, 512);
    gate_kernel<<<(Mrows * 8) / 256, blk, 0, stream>>>(act1, g_w3, g_b3, gate, Mrows);

    // expert stack (per-expert partials folded with gate)
    gemm_wmma_bf16<true,  true,  true ><<<g512, blk, 0, stream>>>(Xbf,  gate, ew1t, e_b1, act0, Mrows, 512, 480, 3840);
    gemm_wmma_bf16<true,  true,  true ><<<g512, blk, 0, stream>>>(act0, gate, ew2t, e_b2, act1, Mrows, 512, 512, 4096);
    gemm_wmma_bf16<true,  false, false><<<g480, blk, 0, stream>>>(act1, gate, ew3t, e_b3, out,  Mrows, 480, 512, 4096);
}
